// TransformerXLModule_3788161155415
// MI455X (gfx1250) — compile-verified, matrix operand order checked by reading, not run on hardware
//
#include <hip/hip_runtime.h>
#include <hip/hip_bf16.h>
#include <cstdint>
#include <cstddef>

typedef __bf16 bf16;
typedef __attribute__((ext_vector_type(16))) __bf16 v16bf;
typedef __attribute__((ext_vector_type(8)))  __bf16 v8bf;
typedef __attribute__((ext_vector_type(8)))  float  v8f;

#define D_MODEL 1024
#define N_HEADS 16
#define D_HEAD  64
#define N_LAYERS 6
#define D_FF    4096
#define QLEN    1024
#define BATCH   8
#define MLEN    512
#define KLEN    1536   // QLEN + MLEN

// ---------------------------------------------------------------- helpers ---
__device__ __forceinline__ v16bf pack16(v8bf lo, v8bf hi) {
  v16bf r;
#pragma unroll
  for (int i = 0; i < 8; ++i) { r[i] = lo[i]; r[8 + i] = hi[i]; }
  return r;
}
__device__ __forceinline__ v8f zero8() {
  v8f z;
#pragma unroll
  for (int i = 0; i < 8; ++i) z[i] = 0.0f;
  return z;
}
__device__ __forceinline__ v8f wmma_bf16(v16bf a, v16bf b, v8f c) {
  // D = A(16x32 bf16) * B(32x16 bf16) + C(16x16 f32)
  return __builtin_amdgcn_wmma_f32_16x16x32_bf16(false, a, false, b,
                                                 (short)0, c, false, false);
}

// ------------------------------------------------------------ tiled GEMM ---
// C[M,N] = A[M,K] * W[N,K]^T   (A, W bf16 row-major, K contiguous)
// Block 128x128, 4 waves (2x2), 64x64 per wave -> 16 WMMAs per wave per
// K-step against 8 LDS fragment loads and one barrier pair.
// optional bias[N], ReLU; writes float C and/or bf16 C.
#define BM 128
#define BN 128
#define BK 32
#define LDA 40     // padded LDS row stride (halves) -> conflict-free b128 reads
#define LDWS 40

__global__ void __launch_bounds__(128)
gemm_bf16_kernel(const bf16* __restrict__ A, const bf16* __restrict__ W,
                 const float* __restrict__ bias, float* __restrict__ Cf,
                 bf16* __restrict__ Ch, int M, int N, int K, int relu)
{
  __shared__ __align__(16) bf16 As[BM * LDA];
  __shared__ __align__(16) bf16 Ws[BN * LDWS];
  const int tid  = threadIdx.x;
  const int lane = tid & 31;
  const int wid  = tid >> 5;
  const int wm = wid >> 1, wn = wid & 1;       // 2x2 wave grid, 64x64 per wave
  const int lr = lane & 15, lg = lane >> 4;
  const int m0 = blockIdx.y * BM;
  const int n0 = blockIdx.x * BN;

  v8f acc[4][4];
#pragma unroll
  for (int i = 0; i < 4; ++i)
#pragma unroll
    for (int j = 0; j < 4; ++j) acc[i][j] = zero8();

  for (int k0 = 0; k0 < K; k0 += BK) {
    __syncthreads();
#pragma unroll
    for (int c4 = 0; c4 < 4; ++c4) {          // stage A tile: 128x32 halves
      int idx = tid + c4 * 128;
      int row = idx >> 2, cc = idx & 3;
      *(v8bf*)&As[row * LDA + cc * 8] =
          *(const v8bf*)&A[(size_t)(m0 + row) * K + k0 + cc * 8];
      __builtin_prefetch(&A[(size_t)(m0 + row) * K + k0 + BK + cc * 8], 0, 1);
    }
#pragma unroll
    for (int c4 = 0; c4 < 4; ++c4) {          // stage W tile: 128x32 halves
      int idx = tid + c4 * 128;
      int row = idx >> 2, cc = idx & 3;
      *(v8bf*)&Ws[row * LDWS + cc * 8] =
          *(const v8bf*)&W[(size_t)(n0 + row) * K + k0 + cc * 8];
      __builtin_prefetch(&W[(size_t)(n0 + row) * K + k0 + BK + cc * 8], 0, 1);
    }
    __syncthreads();

    v16bf bfr[4];
#pragma unroll
    for (int j = 0; j < 4; ++j) {
      const bf16* p = &Ws[(wn * 64 + j * 16 + lr) * LDWS + lg * 8];
      bfr[j] = pack16(*(const v8bf*)p, *(const v8bf*)(p + 16));
    }
#pragma unroll
    for (int i = 0; i < 4; ++i) {
      const bf16* p = &As[(wm * 64 + i * 16 + lr) * LDA + lg * 8];
      v16bf af = pack16(*(const v8bf*)p, *(const v8bf*)(p + 16));
#pragma unroll
      for (int j = 0; j < 4; ++j)
        acc[i][j] = wmma_bf16(af, bfr[j], acc[i][j]);
    }
  }

  // epilogue (C layout: lane lr = column, rows = v + 8*lg)
#pragma unroll
  for (int i = 0; i < 4; ++i)
#pragma unroll
    for (int j = 0; j < 4; ++j) {
      int col = n0 + wn * 64 + j * 16 + lr;
      float bv = bias ? bias[col] : 0.0f;
#pragma unroll
      for (int v = 0; v < 8; ++v) {
        int row = m0 + wm * 64 + i * 16 + v + 8 * lg;
        float val = acc[i][j][v] + bv;
        if (relu) val = fmaxf(val, 0.0f);
        if (Cf) Cf[(size_t)row * N + col] = val;
        if (Ch) Ch[(size_t)row * N + col] = (bf16)val;
      }
    }
}

// ------------------------------------------------------------- LayerNorm ---
// rows r < rows0 read src0, else src1 (gathers concat(mems, h)); bf16 out.
__global__ void ln_kernel(const float* __restrict__ src0, int rows0,
                          const float* __restrict__ src1,
                          const float* __restrict__ g, const float* __restrict__ b,
                          bf16* __restrict__ out)
{
  const int r = blockIdx.x;
  const int t = threadIdx.x;
  const float* x = (r < rows0) ? (src0 + (size_t)r * D_MODEL)
                               : (src1 + (size_t)(r - rows0) * D_MODEL);
  float v[4];
  float s = 0.0f, sq = 0.0f;
#pragma unroll
  for (int i = 0; i < 4; ++i) {
    v[i] = x[t + i * 256];
    s += v[i]; sq += v[i] * v[i];
  }
  __shared__ float sh1[256], sh2[256];
  sh1[t] = s; sh2[t] = sq;
  __syncthreads();
  for (int off = 128; off > 0; off >>= 1) {
    if (t < off) { sh1[t] += sh1[t + off]; sh2[t] += sh2[t + off]; }
    __syncthreads();
  }
  float mean = sh1[0] * (1.0f / D_MODEL);
  float var  = sh2[0] * (1.0f / D_MODEL) - mean * mean;
  float rs = rsqrtf(var + 1e-5f);
  bf16* o = out + (size_t)r * D_MODEL;
#pragma unroll
  for (int i = 0; i < 4; ++i) {
    int c = t + i * 256;
    o[c] = (bf16)((v[i] - mean) * rs * g[c] + b[c]);
  }
}

// ------------------------------------------------------- small elementwise ---
__global__ void cvt_bf16_kernel(bf16* __restrict__ dst,
                                const float* __restrict__ src, size_t n) {
  for (size_t i = (size_t)blockIdx.x * blockDim.x + threadIdx.x; i < n;
       i += (size_t)gridDim.x * blockDim.x)
    dst[i] = (bf16)src[i];
}
__global__ void copy_f32_kernel(float* __restrict__ dst,
                                const float* __restrict__ src, size_t n) {
  for (size_t i = (size_t)blockIdx.x * blockDim.x + threadIdx.x; i < n;
       i += (size_t)gridDim.x * blockDim.x)
    dst[i] = src[i];
}
__global__ void add_inplace_kernel(float* __restrict__ a,
                                   const float* __restrict__ b, size_t n) {
  for (size_t i = (size_t)blockIdx.x * blockDim.x + threadIdx.x; i < n;
       i += (size_t)gridDim.x * blockDim.x)
    a[i] += b[i];
}

// sinusoidal relative positions, pos = [klen-1 .. 0], bf16 out (KLEN x D)
__global__ void relpos_kernel(bf16* __restrict__ out) {
  int idx = blockIdx.x * 256 + threadIdx.x;
  if (idx >= KLEN * D_MODEL) return;
  int r = idx >> 10, c = idx & 1023;
  float pos = (float)(KLEN - 1 - r);
  int ci = (c < 512) ? c : c - 512;
  float inv = __expf(-((float)ci / 512.0f) * 9.210340371976184f); // ln(10000)
  float ang = pos * inv;
  out[idx] = (bf16)((c < 512) ? sinf(ang) : cosf(ang));
}

// q + r_w_bias / q + r_r_bias  ->  (B, H, QLEN, D_HEAD) bf16
__global__ void prep_q_kernel(const bf16* __restrict__ wh,
                              const float* __restrict__ rwb,
                              const float* __restrict__ rrb,
                              bf16* __restrict__ qac, bf16* __restrict__ qbd) {
  size_t idx = (size_t)blockIdx.x * 256 + threadIdx.x;
  if (idx >= (size_t)BATCH * N_HEADS * QLEN * D_HEAD) return;
  int d = (int)(idx & 63);
  size_t t = idx >> 6;
  int i = (int)(t & (QLEN - 1)); t >>= 10;
  int n = (int)(t & (N_HEADS - 1));
  int b = (int)(t >> 4);
  float q = (float)wh[((size_t)(MLEN + i) * BATCH + b) * 3072 + n * 64 + d];
  qac[idx] = (bf16)(q + rwb[n * 64 + d]);
  qbd[idx] = (bf16)(q + rrb[n * 64 + d]);
}

// k, v  ->  (B, H, KLEN, D_HEAD) bf16
__global__ void prep_kv_kernel(const bf16* __restrict__ wh,
                               bf16* __restrict__ ko, bf16* __restrict__ vo) {
  size_t idx = (size_t)blockIdx.x * 256 + threadIdx.x;
  if (idx >= (size_t)BATCH * N_HEADS * KLEN * D_HEAD) return;
  int d = (int)(idx & 63);
  size_t t = idx >> 6;
  int j = (int)(t % KLEN); t /= KLEN;
  int n = (int)(t & (N_HEADS - 1));
  int b = (int)(t >> 4);
  size_t src = ((size_t)j * BATCH + b) * 3072;
  ko[idx] = wh[src + 1024 + n * 64 + d];
  vo[idx] = wh[src + 2048 + n * 64 + d];
}

// --------------------------------------------------- fused flash attention ---
// Block = (b2, n2, 64-query tile), 4 waves x 16 queries. Computes S^T = K*Q^T
// (lane = query column -> in-lane online softmax), P^T tiles are directly
// reusable as WMMA B-frags for O^T = V^T * P^T. Rel-shift slab permutation
// applied to the BD term. Writes attn_vec (QLEN*B, D_MODEL) bf16.
__global__ void __launch_bounds__(128)
attn_kernel(const bf16* __restrict__ qac, const bf16* __restrict__ qbd,
            const bf16* __restrict__ kh, const bf16* __restrict__ vh,
            const bf16* __restrict__ rk, bf16* __restrict__ avec)
{
  const int i0 = blockIdx.x * 64;
  const int n2 = blockIdx.y;
  const int b2 = blockIdx.z;
  const int lane = threadIdx.x & 31;
  const int wave = threadIdx.x >> 5;
  const int lr = lane & 15, lg = lane >> 4;
  const int iw = i0 + wave * 16;

  // _rel_shift slab permutation: out slab (b2,n2) <- x slab (bb, mm-1), 0 if mm==0
  const int S  = b2 * N_HEADS + n2 + BATCH;
  const int bb = S / (N_HEADS + 1);
  const int mm = S % (N_HEADS + 1);
  const bool hasBD = (mm != 0);
  const int nn = hasBD ? (mm - 1) : 0;

  __shared__ __align__(16) bf16 VT[64 * 72];   // V^T tile [d][j], padded
  __shared__ float OS[4][16][68];              // per-wave O scratch

  // Q^T B-frags (lane = query column), for both bias variants, d-blocks 0/1
  const bf16* qa = qac + ((size_t)(b2 * N_HEADS + n2) * QLEN + iw + lr) * D_HEAD;
  const bf16* qb = qbd + ((size_t)(bb * N_HEADS + nn) * QLEN + iw + lr) * D_HEAD;
  v16bf qaf[2], qbf[2];
#pragma unroll
  for (int kb = 0; kb < 2; ++kb) {
    const bf16* p = qa + kb * 32 + lg * 8;
    qaf[kb] = pack16(*(const v8bf*)p, *(const v8bf*)(p + 16));
    const bf16* q = qb + kb * 32 + lg * 8;
    qbf[kb] = pack16(*(const v8bf*)q, *(const v8bf*)(q + 16));
  }

  const bf16* Kb = kh + (size_t)(b2 * N_HEADS + n2) * KLEN * D_HEAD;
  const bf16* Vb = vh + (size_t)(b2 * N_HEADS + n2) * KLEN * D_HEAD;

  v8f o[4];
#pragma unroll
  for (int dt = 0; dt < 4; ++dt) o[dt] = zero8();
  float mx = -3.0e38f, lsum = 0.0f;
  const float scale = 0.125f;                  // 1/sqrt(64)
  const int iq = iw + lr;                      // this lane's query index
  const int ntiles = (i0 + 63 + MLEN) / 64 + 1; // skip fully-masked key tiles

  for (int kt = 0; kt < ntiles; ++kt) {
    const int j0 = kt * 64;
    __syncthreads();
    {   // cooperative transpose-stage of V tile (64 keys x 64 d) -> VT[d][j]
      const int j  = threadIdx.x >> 1;
      const int c0 = (threadIdx.x & 1) * 32;
      const bf16* vr = Vb + (size_t)(j0 + j) * D_HEAD + c0;
#pragma unroll
      for (int q8 = 0; q8 < 4; ++q8) {
        v8bf val = *(const v8bf*)(vr + q8 * 8);
#pragma unroll
        for (int e = 0; e < 8; ++e)
          VT[(c0 + q8 * 8 + e) * 72 + j] = val[e];
      }
    }
    __syncthreads();

    // S^T strip: 4 C-tiles of (16 keys x 16 queries), K-reduce over d=64
    v8f sc[4];
#pragma unroll
    for (int t = 0; t < 4; ++t) {
      v8f a = zero8();
#pragma unroll
      for (int kb = 0; kb < 2; ++kb) {         // AC term
        const bf16* p = Kb + (size_t)(j0 + t * 16 + lr) * D_HEAD + kb * 32 + lg * 8;
        v16bf af = pack16(*(const v8bf*)p, *(const v8bf*)(p + 16));
        a = wmma_bf16(af, qaf[kb], a);
      }
      if (hasBD) {                             // BD term (permuted slab)
#pragma unroll
        for (int kb = 0; kb < 2; ++kb) {
          const bf16* p = rk + (size_t)(j0 + t * 16 + lr) * (N_HEADS * D_HEAD)
                             + nn * D_HEAD + kb * 32 + lg * 8;
          v16bf af = pack16(*(const v8bf*)p, *(const v8bf*)(p + 16));
          a = wmma_bf16(af, qbf[kb], a);
        }
      }
      sc[t] = a;
    }

    // scale + causal mask + online softmax (lane owns one query column)
    float mnew = mx;
#pragma unroll
    for (int t = 0; t < 4; ++t)
#pragma unroll
      for (int v = 0; v < 8; ++v) {
        int j = j0 + t * 16 + v + 8 * lg;
        float s = sc[t][v] * scale;
        if (j > iq + MLEN) s = -1.0e30f;
        sc[t][v] = s;
        mnew = fmaxf(mnew, s);
      }
    mnew = fmaxf(mnew, __shfl_xor(mnew, 16, 32));
    const float alpha = __expf(mx - mnew);
    mx = mnew;
    lsum *= alpha;
#pragma unroll
    for (int dt = 0; dt < 4; ++dt)
#pragma unroll
      for (int v = 0; v < 8; ++v) o[dt][v] *= alpha;

    float ls = 0.0f;
#pragma unroll
    for (int t = 0; t < 4; ++t)
#pragma unroll
      for (int v = 0; v < 8; ++v) {
        float p = __expf(sc[t][v] - mx);
        sc[t][v] = p;
        ls += p;
      }
    lsum += ls + __shfl_xor(ls, 16, 32);

    // O^T += V^T * P^T : P C-frags map 1:1 onto 32x16 B-frags (no shuffles)
#pragma unroll
    for (int kb2 = 0; kb2 < 2; ++kb2) {
      v16bf pf;
#pragma unroll
      for (int e = 0; e < 8; ++e) {
        pf[e]     = (bf16)sc[2 * kb2][e];
        pf[8 + e] = (bf16)sc[2 * kb2 + 1][e];
      }
#pragma unroll
      for (int dt = 0; dt < 4; ++dt) {
        const bf16* p = &VT[(dt * 16 + lr) * 72 + kb2 * 32 + lg * 8];
        v16bf af = pack16(*(const v8bf*)p, *(const v8bf*)(p + 16));
        o[dt] = wmma_bf16(af, pf, o[dt]);
      }
    }
  }

  // normalize + transpose via LDS (same-wave, no barrier needed) + store bf16
  const float inv = 1.0f / lsum;
#pragma unroll
  for (int dt = 0; dt < 4; ++dt)
#pragma unroll
    for (int v = 0; v < 8; ++v)
      OS[wave][lr][dt * 16 + v + 8 * lg] = o[dt][v] * inv;
  {
    const int qq = lane >> 1;
    const int c0 = (lane & 1) * 32;
    bf16* dst = avec + ((size_t)(iw + qq) * BATCH + b2) * D_MODEL + n2 * D_HEAD + c0;
#pragma unroll
    for (int q8 = 0; q8 < 4; ++q8) {
      v8bf val;
#pragma unroll
      for (int e = 0; e < 8; ++e) val[e] = (bf16)OS[wave][qq][c0 + q8 * 8 + e];
      *(v8bf*)(dst + q8 * 8) = val;
    }
  }
}

// ---------------------------------------------------------------- launch ---
extern "C" void kernel_launch(void* const* d_in, const int* in_sizes, int n_in,
                              void* d_out, int out_size, void* d_ws, size_t ws_size,
                              hipStream_t stream)
{
  (void)in_sizes; (void)n_in; (void)out_size; (void)ws_size;
  const float* inputs = (const float*)d_in[0];
  const float* mems   = (const float*)d_in[1];
  const float* qkv_w  = (const float*)d_in[2];
  const float* r_w    = (const float*)d_in[3];
  const float* o_w    = (const float*)d_in[4];
  const float* r_w_b  = (const float*)d_in[5];
  const float* r_r_b  = (const float*)d_in[6];
  const float* aln_g  = (const float*)d_in[7];
  const float* aln_b  = (const float*)d_in[8];
  const float* ff_w1  = (const float*)d_in[9];
  const float* ff_b1  = (const float*)d_in[10];
  const float* ff_w2  = (const float*)d_in[11];
  const float* ff_b2  = (const float*)d_in[12];
  const float* fln_g  = (const float*)d_in[13];
  const float* fln_b  = (const float*)d_in[14];
  float* h = (float*)d_out;                    // residual stream lives in d_out

  // workspace carving (256B aligned)
  char* wp = (char*)d_ws;
  auto carve = [&](size_t bytes) -> void* {
    void* r = (void*)wp; wp += (bytes + 255) & ~(size_t)255; return r;
  };
  bf16* qkvw_h  = (bf16*)carve((size_t)N_LAYERS * 3072 * 1024 * 2);
  bf16* rw_h    = (bf16*)carve((size_t)N_LAYERS * 1024 * 1024 * 2);
  bf16* ow_h    = (bf16*)carve((size_t)N_LAYERS * 1024 * 1024 * 2);
  bf16* ff1w_h  = (bf16*)carve((size_t)N_LAYERS * 4096 * 1024 * 2);
  bf16* ff2w_h  = (bf16*)carve((size_t)N_LAYERS * 1024 * 4096 * 2);
  bf16* relp_h  = (bf16*)carve((size_t)KLEN * D_MODEL * 2);
  bf16* lnout_h = (bf16*)carve((size_t)KLEN * BATCH * D_MODEL * 2);
  bf16* wh_h    = (bf16*)carve((size_t)KLEN * BATCH * 3072 * 2);
  bf16* qac_h   = (bf16*)carve((size_t)BATCH * N_HEADS * QLEN * D_HEAD * 2);
  bf16* qbd_h   = (bf16*)carve((size_t)BATCH * N_HEADS * QLEN * D_HEAD * 2);
  bf16* k_h     = (bf16*)carve((size_t)BATCH * N_HEADS * KLEN * D_HEAD * 2);
  bf16* v_h     = (bf16*)carve((size_t)BATCH * N_HEADS * KLEN * D_HEAD * 2);
  bf16* rk_h    = (bf16*)carve((size_t)KLEN * D_MODEL * 2);
  bf16* avec_h  = (bf16*)carve((size_t)QLEN * BATCH * D_MODEL * 2);
  bf16* ff1a_h  = (bf16*)carve((size_t)QLEN * BATCH * D_FF * 2);
  float* ctmp   = (float*)carve((size_t)QLEN * BATCH * D_MODEL * 4);

  // weights fp32 -> bf16 (recomputed every call; trivially cheap vs compute)
  cvt_bf16_kernel<<<4096, 256, 0, stream>>>(qkvw_h, qkv_w, (size_t)N_LAYERS*3072*1024);
  cvt_bf16_kernel<<<4096, 256, 0, stream>>>(rw_h,   r_w,   (size_t)N_LAYERS*1024*1024);
  cvt_bf16_kernel<<<4096, 256, 0, stream>>>(ow_h,   o_w,   (size_t)N_LAYERS*1024*1024);
  cvt_bf16_kernel<<<4096, 256, 0, stream>>>(ff1w_h, ff_w1, (size_t)N_LAYERS*4096*1024);
  cvt_bf16_kernel<<<4096, 256, 0, stream>>>(ff2w_h, ff_w2, (size_t)N_LAYERS*1024*4096);
  relpos_kernel<<<(KLEN * D_MODEL) / 256, 256, 0, stream>>>(relp_h);
  copy_f32_kernel<<<4096, 256, 0, stream>>>(h, inputs, (size_t)QLEN * BATCH * D_MODEL);

  const size_t nQ  = (size_t)BATCH * N_HEADS * QLEN * D_HEAD;   // 8,388,608
  const size_t nKV = (size_t)BATCH * N_HEADS * KLEN * D_HEAD;   // 12,582,912

  for (int l = 0; l < N_LAYERS; ++l) {
    const float* mems_l = mems + (size_t)l * MLEN * BATCH * D_MODEL;
    // --- attention ---
    ln_kernel<<<KLEN * BATCH, 256, 0, stream>>>(
        mems_l, MLEN * BATCH, h, aln_g + l * D_MODEL, aln_b + l * D_MODEL, lnout_h);
    gemm_bf16_kernel<<<dim3(3072 / BN, (KLEN * BATCH) / BM), 128, 0, stream>>>(
        lnout_h, qkvw_h + (size_t)l * 3072 * 1024, nullptr, nullptr, wh_h,
        KLEN * BATCH, 3072, 1024, 0);
    prep_q_kernel<<<(int)(nQ / 256), 256, 0, stream>>>(wh_h, r_w_b, r_r_b, qac_h, qbd_h);
    prep_kv_kernel<<<(int)(nKV / 256), 256, 0, stream>>>(wh_h, k_h, v_h);
    gemm_bf16_kernel<<<dim3(1024 / BN, KLEN / BM), 128, 0, stream>>>(
        relp_h, rw_h + (size_t)l * 1024 * 1024, nullptr, nullptr, rk_h,
        KLEN, 1024, 1024, 0);
    attn_kernel<<<dim3(QLEN / 64, N_HEADS, BATCH), 128, 0, stream>>>(
        qac_h, qbd_h, k_h, v_h, rk_h, avec_h);
    gemm_bf16_kernel<<<dim3(1024 / BN, (QLEN * BATCH) / BM), 128, 0, stream>>>(
        avec_h, ow_h + (size_t)l * 1024 * 1024, nullptr, ctmp, nullptr,
        QLEN * BATCH, 1024, 1024, 0);
    add_inplace_kernel<<<4096, 256, 0, stream>>>(h, ctmp, (size_t)QLEN * BATCH * D_MODEL);
    // --- FFN ---
    ln_kernel<<<QLEN * BATCH, 256, 0, stream>>>(
        h, QLEN * BATCH, h, fln_g + l * D_MODEL, fln_b + l * D_MODEL, lnout_h);
    gemm_bf16_kernel<<<dim3(D_FF / BN, (QLEN * BATCH) / BM), 128, 0, stream>>>(
        lnout_h, ff1w_h + (size_t)l * D_FF * 1024, ff_b1 + (size_t)l * D_FF,
        nullptr, ff1a_h, QLEN * BATCH, D_FF, 1024, 1);
    gemm_bf16_kernel<<<dim3(1024 / BN, (QLEN * BATCH) / BM), 128, 0, stream>>>(
        ff1a_h, ff2w_h + (size_t)l * 1024 * D_FF, ff_b2 + (size_t)l * 1024,
        ctmp, nullptr, QLEN * BATCH, 1024, D_FF, 0);
    add_inplace_kernel<<<4096, 256, 0, stream>>>(h, ctmp, (size_t)QLEN * BATCH * D_MODEL);
  }
}
